// MambaAdapter_44083544326208
// MI455X (gfx1250) — matrix operand order repaired and checked by reading, not compiled, not run on hardware
//
#include <hip/hip_runtime.h>
#include <hip/hip_bf16.h>
#include <stdint.h>
#include <stddef.h>

// ---------------- types ----------------
typedef __attribute__((ext_vector_type(16))) __bf16 bf16x16;
typedef __attribute__((ext_vector_type(8)))  float  f32x8;
typedef __attribute__((ext_vector_type(4)))  unsigned int u32x4;

#define BATCH   4096
#define DMODEL  512
#define DINNER  1024
#define DSTATE  16
#define DTRANK  32
#define BN_EPS  1e-5f

// ---------------- helpers ----------------
__device__ __forceinline__ unsigned short f32_to_bf16_bits(float f) {
  union { float f; unsigned int u; } v; v.f = f;
  unsigned int u = v.u;
  u += 0x7FFFu + ((u >> 16) & 1u);   // round-to-nearest-even
  return (unsigned short)(u >> 16);
}
__device__ __forceinline__ float sigmoidf_(float x) { return 1.0f / (1.0f + expf(-x)); }
__device__ __forceinline__ float softplusf_(float x) { return (x > 20.0f) ? x : log1pf(expf(x)); }

// ---------------- elementwise kernels ----------------
__global__ void k_f32_to_bf16(const float* __restrict__ src,
                              unsigned short* __restrict__ dst, int n) {
  int i = blockIdx.x * blockDim.x + threadIdx.x;
  if (i < n) dst[i] = f32_to_bf16_bits(src[i]);
}

// h = BN(feat) -> bf16, over BATCH*DMODEL (exact grid)
__global__ void k_bn_to_bf16(const float* __restrict__ feat,
                             const float* __restrict__ g, const float* __restrict__ b,
                             const float* __restrict__ mu, const float* __restrict__ var,
                             unsigned short* __restrict__ h) {
  int i = blockIdx.x * blockDim.x + threadIdx.x;     // [0, BATCH*DMODEL)
  int c = i & (DMODEL - 1);
  float t = (feat[i] - mu[c]) * rsqrtf(var[c] + BN_EPS) * g[c] + b[c];
  h[i] = f32_to_bf16_bits(t);
}

// In-place on xz[B,2*DINNER]: col d -> xc = silu(w3*xi + b), col 1024+d -> silu(z).
// Also write xc as bf16 for the next GEMM.
__global__ void k_act(float* __restrict__ xz, unsigned short* __restrict__ xcb,
                      const float* __restrict__ conv_w, const float* __restrict__ conv_b) {
  int i = blockIdx.x * blockDim.x + threadIdx.x;     // [0, BATCH*DINNER)
  int d = i & (DINNER - 1);
  size_t base = (size_t)(i >> 10) * (2 * DINNER);
  float xi  = xz[base + d];
  float pre = conv_w[d * 4 + 3] * xi + conv_b[d];    // only last tap alive at L==1
  float xc  = pre * sigmoidf_(pre);
  xz[base + d] = xc;
  xcb[i] = f32_to_bf16_bits(xc);
  float zz = xz[base + DINNER + d];
  xz[base + DINNER + d] = zz * sigmoidf_(zz);
}

// Per row: s = dot(Bm, Cm); dt -> bf16
__global__ void k_scs(const float* __restrict__ xdbl, float* __restrict__ s,
                      unsigned short* __restrict__ dtb) {
  int b = blockIdx.x * blockDim.x + threadIdx.x;
  if (b >= BATCH) return;
  const float* row = xdbl + (size_t)b * (DTRANK + 2 * DSTATE);
  float acc = 0.0f;
#pragma unroll
  for (int i = 0; i < DSTATE; ++i) acc += row[DTRANK + i] * row[DTRANK + DSTATE + i];
  s[b] = acc;
#pragma unroll
  for (int i = 0; i < DTRANK; ++i) dtb[(size_t)b * DTRANK + i] = f32_to_bf16_bits(row[i]);
}

// ---------------- WMMA GEMM: C[M,N] = A[M,K](bf16) @ W[N,K]^T(bf16) ----------------
// EPI 0: plain f32 store.  EPI 1: mamba gate -> bf16.  EPI 2: BatchNorm -> f32.
template <int EPI>
__device__ __forceinline__ void store_epi(float* outF, unsigned short* outB,
                                          int m, int n, int N, float v,
                                          const float* p0, const float* p1,
                                          const float* p2, const float* p3) {
  if constexpr (EPI == 0) {
    outF[(size_t)m * N + n] = v;
  } else if constexpr (EPI == 1) {
    // p0 = s[B], p1 = dt_bias[N], p2 = D_param[N], p3 = xz base (xc | silu(z), stride 2*DINNER)
    float delta = softplusf_(v + p1[n]);
    float xc = p3[(size_t)m * (2 * DINNER) + n];
    float sz = p3[(size_t)m * (2 * DINNER) + DINNER + n];
    float y  = (delta * p0[m] + p2[n]) * xc * sz;
    outB[(size_t)m * N + n] = f32_to_bf16_bits(y);
  } else {
    // p0 = gamma, p1 = beta, p2 = mean, p3 = var
    float t = (v - p2[n]) * rsqrtf(p3[n] + BN_EPS) * p0[n] + p1[n];
    outF[(size_t)m * N + n] = t;
  }
}

template <int EPI>
__global__ __launch_bounds__(128, 2) void gemm_bf16_wmma(
    const unsigned short* __restrict__ A, const unsigned short* __restrict__ W,
    float* __restrict__ outF, unsigned short* __restrict__ outB,
    int M, int N, int K,
    const float* __restrict__ p0, const float* __restrict__ p1,
    const float* __restrict__ p2, const float* __restrict__ p3) {
  const int lane = threadIdx.x & 31;
  const int wave = threadIdx.x >> 5;
  const int wm = blockIdx.y * 64 + (wave >> 1) * 32;   // wave tile 32x32, block tile 64x64
  const int wn = blockIdx.x * 64 + (wave & 1) * 32;
  const int r  = lane & 15;
  const int hi = lane >> 4;

  f32x8 c00 = {0,0,0,0,0,0,0,0};
  f32x8 c01 = {0,0,0,0,0,0,0,0};
  f32x8 c10 = {0,0,0,0,0,0,0,0};
  f32x8 c11 = {0,0,0,0,0,0,0,0};

  // A 16x32 bf16 fragment: lane holds row wm+r; elems 0..7 at K = k0 + 8*hi,
  // elems 8..15 at K = k0 + 16 + 8*hi  (ISA 7.12.2, 16-bit A 16x32 layout).
  const unsigned short* a0p = A + (size_t)(wm + r) * K + hi * 8;
  const unsigned short* a1p = a0p + (size_t)16 * K;
  // B 32x16 bf16 fragment from W[N,K] row-major: lane holds column wn+r,
  // elems 0..7 at K = k0 + 8*hi, elems 8..15 at K = k0 + 16 + 8*hi (same as A).
  const unsigned short* b0p = W + (size_t)(wn + r) * K + hi * 8;
  const unsigned short* b1p = b0p + (size_t)16 * K;

  union Frag { u32x4 q[2]; bf16x16 v; };

  for (int k0 = 0; k0 < K; k0 += 32) {
    Frag a0, a1, b0, b1;
    a0.q[0] = *(const u32x4*)(a0p + k0);      a0.q[1] = *(const u32x4*)(a0p + k0 + 16);
    a1.q[0] = *(const u32x4*)(a1p + k0);      a1.q[1] = *(const u32x4*)(a1p + k0 + 16);
    b0.q[0] = *(const u32x4*)(b0p + k0);      b0.q[1] = *(const u32x4*)(b0p + k0 + 16);
    b1.q[0] = *(const u32x4*)(b1p + k0);      b1.q[1] = *(const u32x4*)(b1p + k0 + 16);

    c00 = __builtin_amdgcn_wmma_f32_16x16x32_bf16(false, a0.v, false, b0.v, (short)0, c00, false, false);
    c01 = __builtin_amdgcn_wmma_f32_16x16x32_bf16(false, a0.v, false, b1.v, (short)0, c01, false, false);
    c10 = __builtin_amdgcn_wmma_f32_16x16x32_bf16(false, a1.v, false, b0.v, (short)0, c10, false, false);
    c11 = __builtin_amdgcn_wmma_f32_16x16x32_bf16(false, a1.v, false, b1.v, (short)0, c11, false, false);
  }

  // D layout: VGPR rr -> row rr + 8*hi, col = lane&15
  const int col = lane & 15;
  const int rb  = hi * 8;
#pragma unroll
  for (int rr = 0; rr < 8; ++rr) {
    store_epi<EPI>(outF, outB, wm + rb + rr,      wn + col,      N, c00[rr], p0, p1, p2, p3);
    store_epi<EPI>(outF, outB, wm + rb + rr,      wn + 16 + col, N, c01[rr], p0, p1, p2, p3);
    store_epi<EPI>(outF, outB, wm + 16 + rb + rr, wn + col,      N, c10[rr], p0, p1, p2, p3);
    store_epi<EPI>(outF, outB, wm + 16 + rb + rr, wn + 16 + col, N, c11[rr], p0, p1, p2, p3);
  }
}

// ---------------- workspace layout (bytes) ----------------
static constexpr size_t O_WIN  = 0;                                 // 2 MB  (2048x512 bf16)
static constexpr size_t O_WXP  = (2ull << 20);                      // 128 KB (64x1024 bf16)
static constexpr size_t O_WDT  = (2ull << 20) + (128ull << 10);     // 64 KB  (1024x32 bf16)
static constexpr size_t O_WOUT = (2ull << 20) + (256ull << 10);     // 1 MB   (512x1024 bf16)
static constexpr size_t O_H    = (4ull << 20);                      // 4 MB   (4096x512 bf16)
static constexpr size_t O_XZ   = (8ull << 20);                      // 32 MB  (4096x2048 f32)
static constexpr size_t O_XCB  = (40ull << 20);                     // 8 MB   (4096x1024 bf16)
static constexpr size_t O_XDBL = (48ull << 20);                     // 1 MB   (4096x64 f32)
static constexpr size_t O_DT   = (49ull << 20);                     // 256 KB (4096x32 bf16)
static constexpr size_t O_S    = (49ull << 20) + (512ull << 10);    // 16 KB  (4096 f32)
static constexpr size_t O_YB   = (50ull << 20);                     // 8 MB   (4096x1024 bf16)

extern "C" void kernel_launch(void* const* d_in, const int* in_sizes, int n_in,
                              void* d_out, int out_size, void* d_ws, size_t ws_size,
                              hipStream_t stream) {
  (void)in_sizes; (void)n_in; (void)out_size; (void)ws_size;
  const float* feat    = (const float*)d_in[0];
  const float* bn_g    = (const float*)d_in[1];
  const float* bn_b    = (const float*)d_in[2];
  const float* bn_m    = (const float*)d_in[3];
  const float* bn_v    = (const float*)d_in[4];
  const float* w_in    = (const float*)d_in[5];
  const float* conv_w  = (const float*)d_in[6];
  const float* conv_b  = (const float*)d_in[7];
  const float* w_xp    = (const float*)d_in[8];
  const float* w_dt    = (const float*)d_in[9];
  const float* dt_bias = (const float*)d_in[10];
  // d_in[11] = A_log: dead at L==1 (h0 == 0 => deltaA multiplies zero)
  const float* D_par   = (const float*)d_in[12];
  const float* w_out   = (const float*)d_in[13];
  float* out = (float*)d_out;

  char* ws = (char*)d_ws;
  unsigned short* winb  = (unsigned short*)(ws + O_WIN);
  unsigned short* wxpb  = (unsigned short*)(ws + O_WXP);
  unsigned short* wdtb  = (unsigned short*)(ws + O_WDT);
  unsigned short* woutb = (unsigned short*)(ws + O_WOUT);
  unsigned short* hb    = (unsigned short*)(ws + O_H);
  float*          xz    = (float*)(ws + O_XZ);
  unsigned short* xcb   = (unsigned short*)(ws + O_XCB);
  float*          xdbl  = (float*)(ws + O_XDBL);
  unsigned short* dtb   = (unsigned short*)(ws + O_DT);
  float*          sbuf  = (float*)(ws + O_S);
  unsigned short* yb    = (unsigned short*)(ws + O_YB);

  // 0) weights f32 -> bf16
  k_f32_to_bf16<<<(2 * DINNER * DMODEL + 255) / 256, 256, 0, stream>>>(w_in,  winb, 2 * DINNER * DMODEL);
  k_f32_to_bf16<<<((DTRANK + 2 * DSTATE) * DINNER + 255) / 256, 256, 0, stream>>>(w_xp, wxpb, (DTRANK + 2 * DSTATE) * DINNER);
  k_f32_to_bf16<<<(DINNER * DTRANK + 255) / 256, 256, 0, stream>>>(w_dt,  wdtb, DINNER * DTRANK);
  k_f32_to_bf16<<<(DMODEL * DINNER + 255) / 256, 256, 0, stream>>>(w_out, woutb, DMODEL * DINNER);

  // 1) h = BN(feat) -> bf16
  k_bn_to_bf16<<<(BATCH * DMODEL) / 256, 256, 0, stream>>>(feat, bn_g, bn_b, bn_m, bn_v, hb);

  // 2) xz = h @ W_in^T : [4096,2048], K=512
  gemm_bf16_wmma<0><<<dim3(2 * DINNER / 64, BATCH / 64), 128, 0, stream>>>(
      hb, winb, xz, nullptr, BATCH, 2 * DINNER, DMODEL, nullptr, nullptr, nullptr, nullptr);

  // 3) in-place: xc = silu(conv tap), silu(z); xc also -> bf16
  k_act<<<(BATCH * DINNER) / 256, 256, 0, stream>>>(xz, xcb, conv_w, conv_b);

  // 4) x_dbl = xc @ W_xp^T : [4096,64], K=1024
  gemm_bf16_wmma<0><<<dim3(1, BATCH / 64), 128, 0, stream>>>(
      xcb, wxpb, xdbl, nullptr, BATCH, DTRANK + 2 * DSTATE, DINNER, nullptr, nullptr, nullptr, nullptr);

  // 5) s = Bm . Cm ; dt -> bf16
  k_scs<<<BATCH / 256, 256, 0, stream>>>(xdbl, sbuf, dtb);

  // 6) y = (softplus(dt @ W_dt^T + bias) * s + D) * xc * silu(z)  -> bf16, K=32
  gemm_bf16_wmma<1><<<dim3(DINNER / 64, BATCH / 64), 128, 0, stream>>>(
      dtb, wdtb, nullptr, yb, BATCH, DINNER, DTRANK, sbuf, dt_bias, D_par, xz);

  // 7) out = BN(y @ W_out^T) : [4096,512], K=1024
  gemm_bf16_wmma<2><<<dim3(DMODEL / 64, BATCH / 64), 128, 0, stream>>>(
      yb, woutb, out, nullptr, BATCH, DMODEL, DINNER, bn_g, bn_b, bn_m, bn_v);
}
